// CosineSimCausalTransformer_38491496907193
// MI455X (gfx1250) — matrix operand order repaired and checked by reading, not compiled, not run on hardware
//
#include <hip/hip_runtime.h>
#include <cstddef>
#include <cstdint>

// Problem constants (match reference)
#define BQ 2
#define NSEQ 1024
#define DMODEL 1024
#define NH 16
#define DHEAD 64
#define VOCAB 32000
#define NLAYERS 6

typedef __attribute__((ext_vector_type(16))) __bf16 v16bf;
typedef __attribute__((ext_vector_type(8)))  __bf16 v8bf;
typedef __attribute__((ext_vector_type(8)))  float  v8f;
typedef __attribute__((ext_vector_type(4))) unsigned int u32x4;
typedef __attribute__((ext_vector_type(4))) int  i32x4;
typedef __attribute__((ext_vector_type(8))) int  i32x8;

union BFrag { v16bf v; v8bf h[2]; };

__device__ inline unsigned short f2bf(float f) {
  union { float f; unsigned u; } c; c.f = f;
  unsigned u = c.u;
  unsigned r = (u + 0x7FFFu + ((u >> 16) & 1u)) >> 16;  // round-nearest-even
  return (unsigned short)r;
}

__device__ inline v8f wmma_bf16(v16bf a, v16bf b, v8f c) {
  // D(16x16,f32) = A(16x32,bf16) x B(32x16,bf16) + C
  return __builtin_amdgcn_wmma_f32_16x16x32_bf16(false, a, false, b, (short)0, c,
                                                 false, false);
}

// Low 32 bits of a generic pointer to LDS = LDS byte address (aperture truncation).
__device__ inline unsigned lds_addr_of(const void* p) {
  return (unsigned)(unsigned long long)(uintptr_t)p;
}

// ------------------------------------------------------------------ TDM load
// 2D tile (tile_k x tile_rows) of bf16 from a row-major [rows, k_len] tensor
// into LDS, with hardware padding so each LDS row is (tile_k*2 + pad) bytes.
// pad_int_code: DWORDs-per-pad-interval code (3 = 16 DW = 64B row for tile_k=32,
// 4 = 32 DW = 128B row for tile_k=64). pad_amt_code 3 = 4 DW = 16B pad.
__device__ inline void tdm_load_2d(unsigned lds_off, const void* gptr,
                                   unsigned tile_k, unsigned tile_rows,
                                   unsigned tensor_k, unsigned tensor_rows,
                                   unsigned long long row_stride_elems,
                                   unsigned pad_int_code, unsigned pad_amt_code) {
  const unsigned long long ga = (unsigned long long)(uintptr_t)gptr;
  u32x4 g0;
  g0[0] = 1u;                                  // count=1, user-mode, no gather
  g0[1] = lds_off;                             // lds_addr
  g0[2] = (unsigned)ga;                        // global_addr[31:0]
  g0[3] = (unsigned)((ga >> 32) & 0x01FFFFFFu) | (2u << 30);  // addr[56:32], type=2
  i32x8 g1;
  g1[0] = (int)((1u << 16) |                   // data_size = 2B
                (1u << 20) |                   // pad_enable
                (pad_int_code << 22) | (pad_amt_code << 25));
  g1[1] = (int)((tensor_k & 0xFFFFu) << 16);                               // dim0 lo
  g1[2] = (int)(((tensor_k >> 16) & 0xFFFFu) | ((tensor_rows & 0xFFFFu) << 16));
  g1[3] = (int)(((tensor_rows >> 16) & 0xFFFFu) | ((tile_k & 0xFFFFu) << 16));
  g1[4] = (int)(tile_rows & 0xFFFFu);          // tile_dim1 (tile_dim2 = 0)
  g1[5] = (int)(row_stride_elems & 0xFFFFFFFFull);   // tensor_dim0_stride lo
  g1[6] = (int)((row_stride_elems >> 32) & 0xFFFFull);
  g1[7] = 0;
  i32x4 gz = {0, 0, 0, 0};
#if defined(__clang_major__) && __clang_major__ >= 23
  i32x8 gz8 = {0, 0, 0, 0, 0, 0, 0, 0};
  __builtin_amdgcn_tensor_load_to_lds(g0, g1, gz, gz, gz8, 0);
#else
  __builtin_amdgcn_tensor_load_to_lds(g0, g1, gz, gz, 0);
#endif
}

// ---------------------------------------------------------------- converts
// dst[l][n][k] = bf16(src[l][k][n]) — LDS-tiled transpose-convert so the GEMM
// B operand is pre-transposed ([N,K], contiguous K per output column).
__global__ __launch_bounds__(256) void transpose_conv_kernel(
    const float* __restrict__ src, unsigned short* __restrict__ dst, int K, int N) {
  __shared__ float t[32][33];
  const int tiles_n = N >> 5, tiles_k = K >> 5;
  long tile = blockIdx.x;
  const int nt = (int)(tile % tiles_n); tile /= tiles_n;
  const int kt = (int)(tile % tiles_k);
  const int l  = (int)(tile / tiles_k);
  const float* s = src + (size_t)l * K * N;
  unsigned short* d = dst + (size_t)l * N * K;
  const int tx = threadIdx.x & 31, ty = threadIdx.x >> 5;  // 32 x 8
#pragma unroll
  for (int r = ty; r < 32; r += 8)
    t[r][tx] = s[(size_t)(kt * 32 + r) * N + nt * 32 + tx];
  __syncthreads();
#pragma unroll
  for (int r = ty; r < 32; r += 8)
    d[(size_t)(nt * 32 + r) * K + kt * 32 + tx] = f2bf(t[tx][r]);
}

// ---------------------------------------------------------------- embedding
__global__ void embed_kernel(const int* __restrict__ x, const float* __restrict__ tok,
                             const float* __restrict__ pos, float* __restrict__ h,
                             long total) {
  long i = (long)blockIdx.x * blockDim.x + threadIdx.x;
  if (i >= total) return;
  int d = (int)(i % DMODEL);
  long bn = i / DMODEL;
  int n = (int)(bn % NSEQ);
  h[i] = tok[(long)x[bn] * DMODEL + d] + pos[(long)n * DMODEL + d];
}

// ---------------------------------------------------------------- layernorm
__global__ __launch_bounds__(256) void ln_bf16_kernel(
    const float* __restrict__ X, const float* __restrict__ g,
    const float* __restrict__ bb, unsigned short* __restrict__ Y, int D) {
  __shared__ float red[256];
  const int row = blockIdx.x;
  const float* x = X + (size_t)row * D;
  float s = 0.f, ss = 0.f;
  for (int i = threadIdx.x; i < D; i += 256) { float v = x[i]; s += v; ss += v * v; }
  red[threadIdx.x] = s; __syncthreads();
  for (int st = 128; st > 0; st >>= 1) {
    if ((int)threadIdx.x < st) red[threadIdx.x] += red[threadIdx.x + st];
    __syncthreads();
  }
  const float mean = red[0] / D;
  __syncthreads();
  red[threadIdx.x] = ss; __syncthreads();
  for (int st = 128; st > 0; st >>= 1) {
    if ((int)threadIdx.x < st) red[threadIdx.x] += red[threadIdx.x + st];
    __syncthreads();
  }
  const float var = red[0] / D - mean * mean;
  const float inv = rsqrtf(var + 1e-5f);
  for (int i = threadIdx.x; i < D; i += 256)
    Y[(size_t)row * D + i] = f2bf((x[i] - mean) * inv * g[i] + bb[i]);
}

// ------------------------------------------------- WMMA GEMM with TDM staging
// C[M,N] = act(A[M,K]bf16 * Wt[N,K]^T bf16 + bias) + resid ; fp32/bf16 out.
// 128x64 block tile, 8 waves each owning a 32x32 slab (4 WMMA / K-step),
// double-buffered LDS filled by tensor_load_to_lds issued from wave 0.
__global__ __launch_bounds__(256) void gemm_bf16_wmma_kernel(
    const unsigned short* __restrict__ A, const unsigned short* __restrict__ Wt,
    const float* __restrict__ bias, const float* __restrict__ resid,
    float* __restrict__ Cf, unsigned short* __restrict__ Cb,
    int M, int N, int K, int act_gelu) {
  constexpr int BM = 128, BN = 64, BK = 32, ST = 40;  // 80B padded LDS rows
  __shared__ __align__(16) unsigned short a_sh[2][BM * ST];
  __shared__ __align__(16) unsigned short b_sh[2][BN * ST];
  const int tid = threadIdx.x;
  const int wave = tid >> 5;
  const int lane = tid & 31;
  const int half = lane >> 4;
  const int lr = lane & 15;
  const int ntiles = N / BN;
  const int m0 = (blockIdx.x / ntiles) * BM;
  const int n0 = (blockIdx.x % ntiles) * BN;
  const int msub = (wave & 3) * 32;   // wave's 32 rows
  const int nsub = (wave >> 2) * 32;  // wave's 32 cols

  const unsigned a_lds[2] = {lds_addr_of(&a_sh[0][0]), lds_addr_of(&a_sh[1][0])};
  const unsigned b_lds[2] = {lds_addr_of(&b_sh[0][0]), lds_addr_of(&b_sh[1][0])};
  const unsigned short* Ab = A + (size_t)m0 * K;
  const unsigned short* Bb = Wt + (size_t)n0 * K;

  const v8f vzero = {};
  v8f acc00 = vzero, acc01 = vzero, acc10 = vzero, acc11 = vzero;

  // prologue: TDM-fill buffer 0 (A 128x32 tile, B 64x32 tile)
  if (tid < 32) {
    tdm_load_2d(a_lds[0], Ab, BK, BM, (unsigned)K, (unsigned)BM,
                (unsigned long long)K, 3, 3);
    tdm_load_2d(b_lds[0], Bb, BK, BN, (unsigned)K, (unsigned)BN,
                (unsigned long long)K, 3, 3);
    __builtin_amdgcn_s_wait_tensorcnt((short)0);
  }
  __syncthreads();

  int buf = 0;
  for (int k0 = 0; k0 < K; k0 += BK) {
    const int nb = buf ^ 1;
    const bool more = (k0 + BK) < K;
    if (more && tid < 32) {  // async-prefetch next K tile while we compute
      tdm_load_2d(a_lds[nb], Ab + (k0 + BK), BK, BM, (unsigned)K, (unsigned)BM,
                  (unsigned long long)K, 3, 3);
      tdm_load_2d(b_lds[nb], Bb + (k0 + BK), BK, BN, (unsigned)K, (unsigned)BN,
                  (unsigned long long)K, 3, 3);
    }
    const unsigned short* as = a_sh[buf];
    const unsigned short* bs = b_sh[buf];
    // A fragments: rows msub(+16)+lr ; K pairs {0-7,16-23} / {8-15,24-31}
    BFrag fa0, fa1, fb0, fb1;
    const int ak = half ? 8 : 0;
    const int ar0 = (msub + lr) * ST, ar1 = (msub + 16 + lr) * ST;
    fa0.h[0] = *reinterpret_cast<const v8bf*>(&as[ar0 + ak]);
    fa0.h[1] = *reinterpret_cast<const v8bf*>(&as[ar0 + ak + 16]);
    fa1.h[0] = *reinterpret_cast<const v8bf*>(&as[ar1 + ak]);
    fa1.h[1] = *reinterpret_cast<const v8bf*>(&as[ar1 + ak + 16]);
    // B fragments: cols nsub(+16)+lr ; K pairs {0-7,16-23} / {8-15,24-31}
    const int bko = half ? 8 : 0;
    const int br0 = (nsub + lr) * ST, br1 = (nsub + 16 + lr) * ST;
    fb0.h[0] = *reinterpret_cast<const v8bf*>(&bs[br0 + bko]);
    fb0.h[1] = *reinterpret_cast<const v8bf*>(&bs[br0 + bko + 16]);
    fb1.h[0] = *reinterpret_cast<const v8bf*>(&bs[br1 + bko]);
    fb1.h[1] = *reinterpret_cast<const v8bf*>(&bs[br1 + bko + 16]);
    acc00 = wmma_bf16(fa0.v, fb0.v, acc00);
    acc01 = wmma_bf16(fa0.v, fb1.v, acc01);
    acc10 = wmma_bf16(fa1.v, fb0.v, acc10);
    acc11 = wmma_bf16(fa1.v, fb1.v, acc11);
    if (more && tid < 32) __builtin_amdgcn_s_wait_tensorcnt((short)0);
    __syncthreads();
    buf = nb;
  }

  // C layout: VGPR j -> row  + j + half*8 ; col = lane&15
#pragma unroll
  for (int mi = 0; mi < 2; ++mi) {
#pragma unroll
    for (int ni = 0; ni < 2; ++ni) {
      const v8f cc = mi ? (ni ? acc11 : acc10) : (ni ? acc01 : acc00);
      const int row0 = m0 + msub + mi * 16 + (half ? 8 : 0);
      const int col = n0 + nsub + ni * 16 + lr;
#pragma unroll
      for (int j = 0; j < 8; ++j) {
        const int row = row0 + j;
        float v = cc[j];
        if (bias) v += bias[col];
        if (act_gelu) v = 0.5f * v * (1.0f + erff(v * 0.70710678118f));
        if (resid) v += resid[(size_t)row * N + col];
        if (Cf) Cf[(size_t)row * N + col] = v;
        if (Cb) Cb[(size_t)row * N + col] = f2bf(v);
      }
    }
  }
}

// ----------------------------------------------- q/k l2norm + scale, to bf16
__global__ __launch_bounds__(64) void qkv_prep_kernel(
    const float* __restrict__ qkv, unsigned short* __restrict__ Qo,
    unsigned short* __restrict__ Ko, unsigned short* __restrict__ Vo) {
  __shared__ float red[64];
  const int idx = blockIdx.x;
  const int hh = idx % NH;
  const int bn = idx / NH;
  const int b = bn / NSEQ, n = bn % NSEQ;
  const int d = threadIdx.x;
  const float* base = qkv + (size_t)bn * (3 * NH * DHEAD);
  const float q = base[hh * DHEAD + d];
  const float k = base[NH * DHEAD + hh * DHEAD + d];
  const float v = base[2 * NH * DHEAD + hh * DHEAD + d];
  red[d] = q * q; __syncthreads();
  for (int st = 32; st > 0; st >>= 1) { if (d < st) red[d] += red[d + st]; __syncthreads(); }
  const float qs = q * rsqrtf(red[0] + 1e-12f) * 8.0f;  // SCALE folded into q
  __syncthreads();
  red[d] = k * k; __syncthreads();
  for (int st = 32; st > 0; st >>= 1) { if (d < st) red[d] += red[d + st]; __syncthreads(); }
  const float ks = k * rsqrtf(red[0] + 1e-12f);
  const size_t o = (((size_t)b * NH + hh) * NSEQ + n) * DHEAD + d;
  Qo[o] = f2bf(qs); Ko[o] = f2bf(ks); Vo[o] = f2bf(v);
}

// --------------------------------------------- flash attention, WMMA both ways
__global__ __launch_bounds__(128) void attn_flash_wmma_kernel(
    const unsigned short* __restrict__ Q, const unsigned short* __restrict__ Kn,
    const unsigned short* __restrict__ Vv, unsigned short* __restrict__ O) {
  constexpr int KST = 72, VST = 40, PST = 40;
  __shared__ __align__(16) unsigned short kt[32 * KST];      // [key][dh]
  __shared__ __align__(16) unsigned short vt[64 * VST];      // [dh][key] (transposed)
  __shared__ __align__(16) unsigned short pt[4 * 16 * PST];  // per-wave P pad
  const int nqb = NSEQ / 64;
  const int bh = blockIdx.x / nqb;
  const int qb = blockIdx.x % nqb;
  const int b = bh / NH, hh = bh % NH;
  const int qbase = qb * 64;
  const int tid = threadIdx.x;
  const int wave = tid >> 5, lane = tid & 31;
  const int half = lane >> 4, lr = lane & 15;
  unsigned short* pw = &pt[wave * 16 * PST];

  const size_t headbase = (size_t)bh * NSEQ * DHEAD;
  const int qr = qbase + wave * 16 + lr;

  // preload Q fragments (A-layout), DH=64 -> two K=32 chunks
  BFrag aq[2];
#pragma unroll
  for (int kc = 0; kc < 2; ++kc) {
    const unsigned short* qp = Q + headbase + (size_t)qr * DHEAD + kc * 32 + (half ? 8 : 0);
    aq[kc].h[0] = *reinterpret_cast<const v8bf*>(qp);
    aq[kc].h[1] = *reinterpret_cast<const v8bf*>(qp + 16);
  }

  const v8f vzero = {};
  v8f oacc[4];
#pragma unroll
  for (int t = 0; t < 4; ++t) oacc[t] = vzero;
  float mrow[8], lrow[8];
#pragma unroll
  for (int j = 0; j < 8; ++j) { mrow[j] = -1e30f; lrow[j] = 0.f; }

  const int lkey = tid >> 2;        // cooperative load: key 0..31
  const int ldh = (tid & 3) << 4;   // dh segment 0/16/32/48
  const int nkb = (qbase + 64) >> 5;

  for (int kb = 0; kb < nkb; ++kb) {
    const int kbase = kb << 5;
    const unsigned short* kp = Kn + headbase + (size_t)(kbase + lkey) * DHEAD + ldh;
    *reinterpret_cast<v8bf*>(&kt[lkey * KST + ldh]) = *reinterpret_cast<const v8bf*>(kp);
    *reinterpret_cast<v8bf*>(&kt[lkey * KST + ldh + 8]) = *reinterpret_cast<const v8bf*>(kp + 8);
    const unsigned short* vp = Vv + headbase + (size_t)(kbase + lkey) * DHEAD + ldh;
    v8bf v0 = *reinterpret_cast<const v8bf*>(vp);
    v8bf v1 = *reinterpret_cast<const v8bf*>(vp + 8);
    const unsigned short* v0s = reinterpret_cast<const unsigned short*>(&v0);
    const unsigned short* v1s = reinterpret_cast<const unsigned short*>(&v1);
#pragma unroll
    for (int j = 0; j < 8; ++j) {
      vt[(ldh + j) * VST + lkey] = v0s[j];
      vt[(ldh + 8 + j) * VST + lkey] = v1s[j];
    }
    if (kb + 1 < nkb) {  // global_prefetch_b8 of next K/V tile
      __builtin_prefetch(kp + 32 * DHEAD, 0, 3);
      __builtin_prefetch(vp + 32 * DHEAD, 0, 3);
    }
    __syncthreads();

    // S = Q * K^T : two 16x16 tiles (32 keys), K-dim = DH = 64
    v8f s[2];
#pragma unroll
    for (int ct = 0; ct < 2; ++ct) {
      s[ct] = vzero;
      const int krow = ct * 16 + lr;  // key column for this lane
#pragma unroll
      for (int kc = 0; kc < 2; ++kc) {
        BFrag fb;
        const int off = kc * 32 + half * 8;
        fb.h[0] = *reinterpret_cast<const v8bf*>(&kt[krow * KST + off]);
        fb.h[1] = *reinterpret_cast<const v8bf*>(&kt[krow * KST + off + 16]);
        s[ct] = wmma_bf16(aq[kc].v, fb.v, s[ct]);
      }
      const int col = kbase + krow;
#pragma unroll
      for (int j = 0; j < 8; ++j) {
        const int row = qbase + wave * 16 + j + half * 8;
        if (col > row) s[ct][j] = -1e30f;  // causal mask
      }
    }

    // online softmax across the 32 columns (16-lane half reductions)
    float scale[8];
#pragma unroll
    for (int j = 0; j < 8; ++j) {
      float mx = fmaxf(s[0][j], s[1][j]);
#pragma unroll
      for (int off = 1; off < 16; off <<= 1) mx = fmaxf(mx, __shfl_xor(mx, off, 32));
      const float mn = fmaxf(mrow[j], mx);
      scale[j] = __expf(mrow[j] - mn);
      mrow[j] = mn;
      const float p0 = __expf(s[0][j] - mn);
      const float p1 = __expf(s[1][j] - mn);
      s[0][j] = p0; s[1][j] = p1;
      float sum = p0 + p1;
#pragma unroll
      for (int off = 1; off < 16; off <<= 1) sum += __shfl_xor(sum, off, 32);
      lrow[j] = lrow[j] * scale[j] + sum;
    }
#pragma unroll
    for (int t = 0; t < 4; ++t)
#pragma unroll
      for (int j = 0; j < 8; ++j) oacc[t][j] *= scale[j];

    // re-layout P: C-layout -> LDS pad -> A-layout (wave-private region)
#pragma unroll
    for (int j = 0; j < 8; ++j) {
      const int prow = j + half * 8;
      pw[prow * PST + lr] = f2bf(s[0][j]);
      pw[prow * PST + 16 + lr] = f2bf(s[1][j]);
    }
    BFrag fp;
    const int pk = half * 8;
    fp.h[0] = *reinterpret_cast<const v8bf*>(&pw[lr * PST + pk]);
    fp.h[1] = *reinterpret_cast<const v8bf*>(&pw[lr * PST + pk + 16]);

    // O(16x64) += P(16x32) * V(32x64) : 4 WMMAs over dh subtiles
#pragma unroll
    for (int t = 0; t < 4; ++t) {
      BFrag fv;
      const int dh = t * 16 + lr;
      fv.h[0] = *reinterpret_cast<const v8bf*>(&vt[dh * VST + half * 8]);
      fv.h[1] = *reinterpret_cast<const v8bf*>(&vt[dh * VST + half * 8 + 16]);
      oacc[t] = wmma_bf16(fp.v, fv.v, oacc[t]);
    }
    __syncthreads();
  }

  // normalize rows and store as [B, N, H*DH] bf16 for the out-projection GEMM
#pragma unroll
  for (int t = 0; t < 4; ++t) {
#pragma unroll
    for (int j = 0; j < 8; ++j) {
      const int row = qbase + wave * 16 + j + half * 8;
      const float val = oacc[t][j] / fmaxf(lrow[j], 1e-20f);
      O[((size_t)b * NSEQ + row) * (NH * DHEAD) + hh * DHEAD + t * 16 + lr] = f2bf(val);
    }
  }
}

// ================================================================ host side
extern "C" void kernel_launch(void* const* d_in, const int* in_sizes, int n_in,
                              void* d_out, int out_size, void* d_ws, size_t ws_size,
                              hipStream_t stream) {
  (void)in_sizes; (void)n_in; (void)out_size; (void)ws_size;
  const int*   x       = (const int*)  d_in[0];
  const float* tok     = (const float*)d_in[1];
  const float* pos     = (const float*)d_in[2];
  const float* ln1_g   = (const float*)d_in[3];
  const float* ln1_b   = (const float*)d_in[4];
  const float* w_qkv   = (const float*)d_in[5];
  const float* w_out   = (const float*)d_in[6];
  const float* ln2_g   = (const float*)d_in[7];
  const float* ln2_b   = (const float*)d_in[8];
  const float* w1      = (const float*)d_in[9];
  const float* b1      = (const float*)d_in[10];
  const float* w2      = (const float*)d_in[11];
  const float* b2      = (const float*)d_in[12];
  const float* lnf_g   = (const float*)d_in[13];
  const float* lnf_b   = (const float*)d_in[14];
  const float* w_logit = (const float*)d_in[15];
  const float* b_logit = (const float*)d_in[16];
  float* out = (float*)d_out;

  char* p = (char*)d_ws;
  auto take = [&](size_t bytes) {
    char* r = p;
    p += (bytes + 255) & ~(size_t)255;
    return r;
  };
  // all weights stored TRANSPOSED: [N, K] bf16
  unsigned short* wqkv_bf = (unsigned short*)take((size_t)NLAYERS * DMODEL * 3072 * 2);
  unsigned short* wout_bf = (unsigned short*)take((size_t)NLAYERS * DMODEL * DMODEL * 2);
  unsigned short* w1_bf   = (unsigned short*)take((size_t)NLAYERS * DMODEL * 4096 * 2);
  unsigned short* w2_bf   = (unsigned short*)take((size_t)NLAYERS * 4096 * DMODEL * 2);
  unsigned short* wlog_bf = (unsigned short*)take((size_t)DMODEL * VOCAB * 2);
  float*          hbuf    = (float*)take((size_t)BQ * NSEQ * DMODEL * 4);
  unsigned short* xn_bf   = (unsigned short*)take((size_t)BQ * NSEQ * DMODEL * 2);
  float*          qkv_f   = (float*)take((size_t)BQ * NSEQ * 3072 * 4);
  unsigned short* q_bf    = (unsigned short*)take((size_t)BQ * NH * NSEQ * DHEAD * 2);
  unsigned short* k_bf    = (unsigned short*)take((size_t)BQ * NH * NSEQ * DHEAD * 2);
  unsigned short* v_bf    = (unsigned short*)take((size_t)BQ * NH * NSEQ * DHEAD * 2);
  unsigned short* o_bf    = (unsigned short*)take((size_t)BQ * NSEQ * DMODEL * 2);
  unsigned short* ffn_bf  = (unsigned short*)take((size_t)BQ * NSEQ * 4096 * 2);

  const int M = BQ * NSEQ;  // 2048 token rows

  // one-shot transpose-converts (deterministic each call): Wt[l][n][k]
  transpose_conv_kernel<<<NLAYERS * (DMODEL / 32) * (3072 / 32), 256, 0, stream>>>(
      w_qkv, wqkv_bf, DMODEL, 3072);
  transpose_conv_kernel<<<NLAYERS * (DMODEL / 32) * (DMODEL / 32), 256, 0, stream>>>(
      w_out, wout_bf, DMODEL, DMODEL);
  transpose_conv_kernel<<<NLAYERS * (DMODEL / 32) * (4096 / 32), 256, 0, stream>>>(
      w1, w1_bf, DMODEL, 4096);
  transpose_conv_kernel<<<NLAYERS * (4096 / 32) * (DMODEL / 32), 256, 0, stream>>>(
      w2, w2_bf, 4096, DMODEL);
  transpose_conv_kernel<<<(DMODEL / 32) * (VOCAB / 32), 256, 0, stream>>>(
      w_logit, wlog_bf, DMODEL, VOCAB);

  embed_kernel<<<((long)M * DMODEL + 255) / 256, 256, 0, stream>>>(
      x, tok, pos, hbuf, (long)M * DMODEL);

  for (int l = 0; l < NLAYERS; ++l) {
    ln_bf16_kernel<<<M, 256, 0, stream>>>(hbuf, ln1_g + (size_t)l * DMODEL,
                                          ln1_b + (size_t)l * DMODEL, xn_bf, DMODEL);
    gemm_bf16_wmma_kernel<<<(M / 128) * (3072 / 64), 256, 0, stream>>>(
        xn_bf, wqkv_bf + (size_t)l * DMODEL * 3072, nullptr, nullptr, qkv_f, nullptr,
        M, 3072, DMODEL, 0);
    qkv_prep_kernel<<<M * NH, 64, 0, stream>>>(qkv_f, q_bf, k_bf, v_bf);
    attn_flash_wmma_kernel<<<BQ * NH * (NSEQ / 64), 128, 0, stream>>>(q_bf, k_bf, v_bf, o_bf);
    gemm_bf16_wmma_kernel<<<(M / 128) * (DMODEL / 64), 256, 0, stream>>>(
        o_bf, wout_bf + (size_t)l * DMODEL * DMODEL, nullptr, hbuf, hbuf, nullptr,
        M, DMODEL, DMODEL, 0);
    ln_bf16_kernel<<<M, 256, 0, stream>>>(hbuf, ln2_g + (size_t)l * DMODEL,
                                          ln2_b + (size_t)l * DMODEL, xn_bf, DMODEL);
    gemm_bf16_wmma_kernel<<<(M / 128) * (4096 / 64), 256, 0, stream>>>(
        xn_bf, w1_bf + (size_t)l * DMODEL * 4096, b1 + (size_t)l * 4096, nullptr,
        nullptr, ffn_bf, M, 4096, DMODEL, 1);
    gemm_bf16_wmma_kernel<<<(M / 128) * (DMODEL / 64), 256, 0, stream>>>(
        ffn_bf, w2_bf + (size_t)l * 4096 * DMODEL, b2 + (size_t)l * DMODEL, hbuf,
        hbuf, nullptr, M, DMODEL, 4096, 0);
  }

  ln_bf16_kernel<<<M, 256, 0, stream>>>(hbuf, lnf_g, lnf_b, xn_bf, DMODEL);
  gemm_bf16_wmma_kernel<<<(M / 128) * (VOCAB / 64), 256, 0, stream>>>(
      xn_bf, wlog_bf, b_logit, nullptr, out, nullptr, M, VOCAB, DMODEL, 0);
}
